// FastMambaBlock_29257317220509
// MI455X (gfx1250) — matrix operand order repaired and checked by reading, not compiled, not run on hardware
//
#include <hip/hip_runtime.h>
#include <hip/hip_bf16.h>

// ---------------------------------------------------------------------------
// FastMambaBlock for MI455X (gfx1250, wave32, WMMA)
//   B=4, L=2048, D_MODEL=1024, D_INNER=2048, D_STATE=16, D_CONV=4, DT_RANK=128
// All GEMMs: v_wmma_f32_16x16x32_bf16, bf16 operands pre-converted, K-loop
// branch-free (pure vmem + wmma; no EXEC manipulation anywhere near WMMA).
// ---------------------------------------------------------------------------

#define BB      4
#define LL      2048
#define DMODEL  1024
#define DINNER  2048
#define DSTATE  16
#define DTRANK  128
#define MTOK    (BB * LL)               // 8192 token rows
#define XPROJ_N (DTRANK + 2 * DSTATE)   // 160

typedef __attribute__((ext_vector_type(16))) __bf16 v16bf;
typedef __attribute__((ext_vector_type(8)))  __bf16 bf16x8;
typedef __attribute__((ext_vector_type(8)))  float  v8f;
typedef __attribute__((ext_vector_type(4)))  float  f32x4;

// float -> bf16, round-to-nearest-even
__device__ __forceinline__ __bf16 f2bf(float f) {
    unsigned int u = __builtin_bit_cast(unsigned int, f);
    u += 0x7FFFu + ((u >> 16) & 1u);
    unsigned short h = (unsigned short)(u >> 16);
    return __builtin_bit_cast(__bf16, h);
}

// A-matrix fragment (16x32 bf16, MxK), ISA layout:
//   elements 0..7  = K kb+0..7  (p[0..7]), 8..15 = K kb+16..23 (p[16..23])
__device__ __forceinline__ v16bf load_a_bf(const __bf16* __restrict__ p) {
    bf16x8 lo = *(const bf16x8*)(p);
    bf16x8 hi = *(const bf16x8*)(p + 16);
    return __builtin_shufflevector(lo, hi, 0, 1, 2, 3, 4, 5, 6, 7,
                                   8, 9, 10, 11, 12, 13, 14, 15);
}

// B-matrix fragment (32x16 bf16, KxN): same K layout as A (p[0..7], p[16..23]).
__device__ __forceinline__ v16bf load_b_bf(const __bf16* __restrict__ p) {
    bf16x8 lo = *(const bf16x8*)(p);
    bf16x8 hi = *(const bf16x8*)(p + 16);
    return __builtin_shufflevector(lo, hi, 0, 1, 2, 3, 4, 5, 6, 7,
                                   8, 9, 10, 11, 12, 13, 14, 15);
}

__device__ __forceinline__ float softplus_f(float x) {
    return fmaxf(x, 0.f) + log1pf(__expf(-fabsf(x)));
}

// ---------------------------------------------------------------------------
// f32 -> bf16 bulk convert (8 elems/thread). n must be a multiple of 8.
// ---------------------------------------------------------------------------
__global__ __launch_bounds__(256) void cvt_f32_bf16(
    const float* __restrict__ src, __bf16* __restrict__ dst, long n) {
    long i = ((long)blockIdx.x * 256 + threadIdx.x) * 8;
    if (i >= n) return;
    f32x4 v0 = *(const f32x4*)(src + i);
    f32x4 v1 = *(const f32x4*)(src + i + 4);
    bf16x8 o;
#pragma unroll
    for (int j = 0; j < 4; ++j) { o[j] = f2bf(v0[j]); o[4 + j] = f2bf(v1[j]); }
    *(bf16x8*)(dst + i) = o;
}

// ---------------------------------------------------------------------------
// GEMM: C[M,N] = A[M,K(lda)] * W[N,K]^T (+ epilogue), bf16 in / f32 accum.
//   EPI 0: += bias    EPI 1: none    EPI 2: softplus(acc + bias)
// Block = 256 threads = 8 waves as 4(M)x2(N); each wave owns a 32x64 tile
// (2 A-frags x 4 B-frags -> 8 WMMAs per 32-K step).
// NGUARD=false: N % 128 == 0, zero guards anywhere.
// NGUARD=true : out-of-range W tiles are CLAMPED (loads stay in-bounds,
//               results discarded), only stores are guarded -> the K-loop is
//               always branch-free and EXEC stays all-ones at every WMMA.
// Optional bf16 copy of columns [0, nbf) into Cbf (row stride ldcb).
// ---------------------------------------------------------------------------
template <int EPI, bool NGUARD>
__global__ __launch_bounds__(256) void gemm_wmma_bf16(
    const __bf16* __restrict__ A, int lda,
    const __bf16* __restrict__ W,
    const float* __restrict__ bias,
    float* __restrict__ C, int ldc,
    __bf16* __restrict__ Cbf, int ldcb, int nbf,
    int M, int N, int K) {
    const int lane = threadIdx.x & 31;
    const int wave = threadIdx.x >> 5;
    const int wm   = wave >> 1;            // 0..3
    const int wn   = wave & 1;             // 0..1
    const int l15  = lane & 15;
    const int hi2  = lane >> 4;            // 0 or 1

    const int m0 = blockIdx.x * 128 + wm * 32;
    const int n0 = blockIdx.y * 128 + wn * 64;

    const __bf16* Ap0 = A + (size_t)(m0 + l15) * lda + 8 * hi2;
    const __bf16* Ap1 = A + (size_t)(m0 + 16 + l15) * lda + 8 * hi2;

    const __bf16* Wp[4];
#pragma unroll
    for (int t = 0; t < 4; ++t) {
        int n = n0 + t * 16;
        if (NGUARD) n = (n < N) ? n : (N - 16);   // clamp: loads in-bounds
        Wp[t] = W + (size_t)(n + l15) * K + 8 * hi2;
    }

    v8f acc0[4] = {};
    v8f acc1[4] = {};

    for (int k = 0; k < K; k += 32) {
        v16bf a0 = load_a_bf(Ap0 + k);
        v16bf a1 = load_a_bf(Ap1 + k);
        v16bf b0 = load_b_bf(Wp[0] + k);
        v16bf b1 = load_b_bf(Wp[1] + k);
        v16bf b2 = load_b_bf(Wp[2] + k);
        v16bf b3 = load_b_bf(Wp[3] + k);
        acc0[0] = __builtin_amdgcn_wmma_f32_16x16x32_bf16(false, a0, false, b0, (short)0, acc0[0], false, false);
        acc1[0] = __builtin_amdgcn_wmma_f32_16x16x32_bf16(false, a1, false, b0, (short)0, acc1[0], false, false);
        acc0[1] = __builtin_amdgcn_wmma_f32_16x16x32_bf16(false, a0, false, b1, (short)0, acc0[1], false, false);
        acc1[1] = __builtin_amdgcn_wmma_f32_16x16x32_bf16(false, a1, false, b1, (short)0, acc1[1], false, false);
        acc0[2] = __builtin_amdgcn_wmma_f32_16x16x32_bf16(false, a0, false, b2, (short)0, acc0[2], false, false);
        acc1[2] = __builtin_amdgcn_wmma_f32_16x16x32_bf16(false, a1, false, b2, (short)0, acc1[2], false, false);
        acc0[3] = __builtin_amdgcn_wmma_f32_16x16x32_bf16(false, a0, false, b3, (short)0, acc0[3], false, false);
        acc1[3] = __builtin_amdgcn_wmma_f32_16x16x32_bf16(false, a1, false, b3, (short)0, acc1[3], false, false);
    }

    // C/D layout: lane(0-15) col=n0+16t+lane, rows +0..7; lane(16-31) rows +8..15
#pragma unroll
    for (int t = 0; t < 4; ++t) {
        if (NGUARD && (n0 + t * 16 >= N)) continue;   // store guard only
        const int n = n0 + t * 16 + l15;
        float bv = 0.f;
        if (EPI != 1) bv = bias[n];
        const bool dobf = (Cbf != nullptr) && (n0 + t * 16 + 16 <= nbf);
#pragma unroll
        for (int half = 0; half < 2; ++half) {
            const int  mrow = m0 + half * 16 + hi2 * 8;
            const v8f& acc  = half ? acc1[t] : acc0[t];
#pragma unroll
            for (int r = 0; r < 8; ++r) {
                float v = acc[r] + bv;
                if (EPI == 2) v = softplus_f(v);
                C[(size_t)(mrow + r) * ldc + n] = v;
                if (dobf) Cbf[(size_t)(mrow + r) * ldcb + n] = f2bf(v);
            }
        }
    }
}

// ---------------------------------------------------------------------------
// Depthwise conv1d, k=4, padding (1,2): out[l] = b + sum_k w[k]*in[l-1+k]
// xr: [MTOK,4096] (xc_raw cols 0..2047). Writes fp32 xc (scan) + bf16 (GEMM).
// ---------------------------------------------------------------------------
__global__ __launch_bounds__(256) void conv_dw_kernel(
    const float* __restrict__ xr,
    const float* __restrict__ conv_w,   // [DINNER, 4]
    const float* __restrict__ conv_b,   // [DINNER]
    float* __restrict__ xc,
    __bf16* __restrict__ xch) {
    const long idx = (long)blockIdx.x * 256 + threadIdx.x; // over MTOK*DINNER
    const int d   = (int)(idx & (DINNER - 1));
    const int row = (int)(idx >> 11);   // b*LL + l
    const int l   = row & (LL - 1);

    const float* w = conv_w + d * 4;
    float acc = conv_b[d];
#pragma unroll
    for (int k = 0; k < 4; ++k) {
        int ls = l + k - 1;
        if (ls >= 0 && ls < LL)
            acc += w[k] * xr[(size_t)(row + k - 1) * 4096 + d];
    }
    xc[(size_t)row * DINNER + d]  = acc;
    xch[(size_t)row * DINNER + d] = f2bf(acc);
}

// ---------------------------------------------------------------------------
// Selective scan + D skip + SiLU gate, fused. One thread per (b,d);
// 16 states in registers; B_ssm/C staged in LDS per 64-step chunk.
// Writes gated y directly as bf16 for the out_proj GEMM.
// ---------------------------------------------------------------------------
#define LCHUNK 64
__global__ __launch_bounds__(256) void scan_kernel(
    const float* __restrict__ xc,     // [MTOK, DINNER] u
    const float* __restrict__ xr,     // [MTOK, 4096]  delta | res
    const float* __restrict__ xdbl,   // [MTOK, 160]   B at 128, C at 144
    const float* __restrict__ A_log,  // [DINNER, 16]
    const float* __restrict__ Dp,     // [DINNER]
    __bf16* __restrict__ yh) {        // [MTOK, DINNER]
    __shared__ float Bs[LCHUNK * DSTATE];
    __shared__ float Cs[LCHUNK * DSTATE];

    const int b = blockIdx.y;
    const int d = blockIdx.x * 256 + threadIdx.x;

    float a[DSTATE];
#pragma unroll
    for (int n = 0; n < DSTATE; ++n)
        a[n] = -__expf(A_log[(size_t)d * DSTATE + n]);
    const float dscale = Dp[d];

    float h[DSTATE];
#pragma unroll
    for (int n = 0; n < DSTATE; ++n) h[n] = 0.f;

    for (int l0 = 0; l0 < LL; l0 += LCHUNK) {
        __syncthreads();
        for (int i = threadIdx.x; i < LCHUNK * DSTATE; i += 256) {
            const int ll = i >> 4, nn = i & 15;
            const float* row = xdbl + (size_t)(b * LL + l0 + ll) * XPROJ_N;
            Bs[i] = row[DTRANK + nn];
            Cs[i] = row[DTRANK + DSTATE + nn];
        }
        __syncthreads();

        for (int li = 0; li < LCHUNK; ++li) {
            const int l = l0 + li;
            const size_t rbase = ((size_t)b * LL + l) * 4096;
            const float dlt = xr[rbase + d];
            const float u   = xc[((size_t)b * LL + l) * DINNER + d];
            const float res = xr[rbase + 2048 + d];
            const float du  = dlt * u;
            float y = 0.f;
#pragma unroll
            for (int n = 0; n < DSTATE; ++n) {
                const float dA = __expf(dlt * a[n]);
                h[n] = dA * h[n] + du * Bs[li * DSTATE + n];
                y += h[n] * Cs[li * DSTATE + n];
            }
            y += u * dscale;
            const float gate = res / (1.f + __expf(-res));  // silu(res)
            yh[((size_t)b * LL + l) * DINNER + d] = f2bf(y * gate);
        }
    }
}

// ---------------------------------------------------------------------------
// Workspace layout (~292 MB total)
// ---------------------------------------------------------------------------
extern "C" void kernel_launch(void* const* d_in, const int* in_sizes, int n_in,
                              void* d_out, int out_size, void* d_ws, size_t ws_size,
                              hipStream_t stream) {
    (void)in_sizes; (void)n_in; (void)out_size; (void)ws_size;

    const float* x        = (const float*)d_in[0];
    const float* in_w     = (const float*)d_in[1];
    const float* in_b     = (const float*)d_in[2];
    const float* conv_w   = (const float*)d_in[3];
    const float* conv_b   = (const float*)d_in[4];
    const float* xproj_w  = (const float*)d_in[5];
    const float* dt_w     = (const float*)d_in[6];
    const float* dt_b     = (const float*)d_in[7];
    const float* A_log    = (const float*)d_in[8];
    const float* Dv       = (const float*)d_in[9];
    const float* out_w    = (const float*)d_in[10];
    const float* out_b    = (const float*)d_in[11];
    float* out            = (float*)d_out;

    char* ws = (char*)d_ws;
    size_t off = 0;
    auto alloc = [&](size_t bytes) { char* p = ws + off; off += (bytes + 255) & ~(size_t)255; return p; };

    float*  XR   = (float*) alloc((size_t)MTOK * 4096   * 4); // xc_raw|res -> delta|res
    float*  XC   = (float*) alloc((size_t)MTOK * DINNER * 4); // conv out (u)
    float*  XDBL = (float*) alloc((size_t)MTOK * XPROJ_N* 4); // x_dbl (B,C read fp32)
    __bf16* Xh   = (__bf16*)alloc((size_t)MTOK * DMODEL * 2); // bf16(x)
    __bf16* XCh  = (__bf16*)alloc((size_t)MTOK * DINNER * 2); // bf16(xc)
    __bf16* XDH  = (__bf16*)alloc((size_t)MTOK * DTRANK * 2); // bf16(x_dbl[:, :128])
    __bf16* Yh   = (__bf16*)alloc((size_t)MTOK * DINNER * 2); // bf16(gated y)
    __bf16* Wih  = (__bf16*)alloc((size_t)(2 * DINNER) * DMODEL * 2);
    __bf16* Wxh  = (__bf16*)alloc((size_t)XPROJ_N * DINNER * 2);
    __bf16* Wdh  = (__bf16*)alloc((size_t)DINNER * DTRANK * 2);
    __bf16* Woh  = (__bf16*)alloc((size_t)DMODEL * DINNER * 2);

    const dim3 blk(256);
    auto cgrid = [](long n) { return (unsigned)((n / 8 + 255) / 256); };

    // 0) one-time fp32 -> bf16 conversions (weights L2-resident afterwards)
    cvt_f32_bf16<<<cgrid((long)MTOK * DMODEL), blk, 0, stream>>>(x, Xh, (long)MTOK * DMODEL);
    cvt_f32_bf16<<<cgrid((long)2 * DINNER * DMODEL), blk, 0, stream>>>(in_w, Wih, (long)2 * DINNER * DMODEL);
    cvt_f32_bf16<<<cgrid((long)XPROJ_N * DINNER), blk, 0, stream>>>(xproj_w, Wxh, (long)XPROJ_N * DINNER);
    cvt_f32_bf16<<<cgrid((long)DINNER * DTRANK), blk, 0, stream>>>(dt_w, Wdh, (long)DINNER * DTRANK);
    cvt_f32_bf16<<<cgrid((long)DMODEL * DINNER), blk, 0, stream>>>(out_w, Woh, (long)DMODEL * DINNER);

    const int mgrid = MTOK / 128;   // 64

    // 1) in_proj: XR = x @ in_w^T + in_b          [8192,4096]
    gemm_wmma_bf16<0, false><<<dim3(mgrid, 4096 / 128), blk, 0, stream>>>(
        Xh, DMODEL, Wih, in_b, XR, 4096, nullptr, 0, 0, MTOK, 2 * DINNER, DMODEL);

    // 2) depthwise conv over L -> XC (fp32) + XCh (bf16)
    conv_dw_kernel<<<(MTOK * DINNER) / 256, blk, 0, stream>>>(XR, conv_w, conv_b, XC, XCh);

    // 3) x_proj: XDBL = xc @ xproj_w^T            [8192,160] (+bf16 dt cols)
    gemm_wmma_bf16<1, true><<<dim3(mgrid, (XPROJ_N + 127) / 128), blk, 0, stream>>>(
        XCh, DINNER, Wxh, nullptr, XDBL, XPROJ_N, XDH, DTRANK, DTRANK,
        MTOK, XPROJ_N, DINNER);

    // 4) dt_proj + softplus: delta -> XR cols 0..2047 (ldc=4096)
    gemm_wmma_bf16<2, false><<<dim3(mgrid, DINNER / 128), blk, 0, stream>>>(
        XDH, DTRANK, Wdh, dt_b, XR, 4096, nullptr, 0, 0, MTOK, DINNER, DTRANK);

    // 5) selective scan + D skip + silu(res) gate -> Yh (bf16)
    scan_kernel<<<dim3(DINNER / 256, BB), blk, 0, stream>>>(XC, XR, XDBL, A_log, Dv, Yh);

    // 6) out_proj: out = y @ out_w^T + out_b      [8192,1024]
    gemm_wmma_bf16<0, false><<<dim3(mgrid, DMODEL / 128), blk, 0, stream>>>(
        Yh, DINNER, Woh, out_b, out, DMODEL, nullptr, 0, 0, MTOK, DMODEL, DINNER);
}
